// BertSelfAttention_21715354649137
// MI455X (gfx1250) — matrix operand order repaired and checked by reading, not compiled, not run on hardware
//
#include <hip/hip_runtime.h>
#include <hip/hip_bf16.h>

// ---------------------------------------------------------------------------
// Types for gfx1250 WMMA (wave32): v16 bf16 A/B fragments, v8 f32 accumulator
// ---------------------------------------------------------------------------
typedef __attribute__((ext_vector_type(16))) __bf16 bf16x16;
typedef __attribute__((ext_vector_type(8)))  __bf16 bf16x8;
typedef __attribute__((ext_vector_type(8)))  float  f32x8;

union FragU { bf16x16 v16; bf16x8 v8[2]; };

#define B_    2
#define S_    2048
#define HID_  768
#define NH_   12
#define HD_   64
#define EROWS 4095   // 2*MAX_POS - 1

// Wave-local LDS ordering fence: all prior ds stores complete before any
// later ds load is issued; "memory" stops compiler reordering across it.
__device__ __forceinline__ void lds_fence_wave() {
  asm volatile("s_wait_dscnt 0x0" ::: "memory");
}

// ---------------------------------------------------------------------------
// Fragment loaders.
// A-matrix 16x32 bf16 layout: lane m (and m+16) hold row m.
//   lane (m, h=lane>>4): K = [k0+16h, +16) contiguous.
// B-matrix 32x16 bf16 (column-major load of source rows, same k order as load_a):
//   lane (n, h): column n = source row n, K = [k0+16h, +16) contiguous.
// ---------------------------------------------------------------------------
__device__ __forceinline__ bf16x16 load_a(const __hip_bfloat16* base, int ld, int k0) {
  int lane = threadIdx.x & 31;
  int m = lane & 15, h = lane >> 4;
  const __hip_bfloat16* p = base + (size_t)m * ld + k0 + 16 * h;
  FragU u;
  u.v8[0] = *reinterpret_cast<const bf16x8*>(p);
  u.v8[1] = *reinterpret_cast<const bf16x8*>(p + 8);
  return u.v16;
}

__device__ __forceinline__ bf16x16 load_b(const __hip_bfloat16* base, int ld, int k0) {
  int lane = threadIdx.x & 31;
  int n = lane & 15, h = lane >> 4;
  const __hip_bfloat16* p = base + (size_t)n * ld + k0 + 16 * h;
  FragU u;
  u.v8[0] = *reinterpret_cast<const bf16x8*>(p);
  u.v8[1] = *reinterpret_cast<const bf16x8*>(p + 8);
  return u.v16;
}

// B fragment from dist_emb window: column j = E row (j0 + jt*16 + n), clamped.
__device__ __forceinline__ bf16x16 load_bE(const __hip_bfloat16* eb, int j0, int jt, int k0) {
  int lane = threadIdx.x & 31;
  int n = lane & 15, h = lane >> 4;
  int row = j0 + jt * 16 + n;
  row = row < 0 ? 0 : (row > EROWS - 1 ? EROWS - 1 : row);
  const __hip_bfloat16* p = eb + (size_t)row * HD_ + k0 + 16 * h;
  FragU u;
  u.v8[0] = *reinterpret_cast<const bf16x8*>(p);
  u.v8[1] = *reinterpret_cast<const bf16x8*>(p + 8);
  return u.v16;
}

__device__ __forceinline__ f32x8 wmma_bf16(bf16x16 a, bf16x16 b, f32x8 c) {
  return __builtin_amdgcn_wmma_f32_16x16x32_bf16(false, a, false, b, (short)0, c, false, false);
}

// ---------------------------------------------------------------------------
// Kernel 0: f32 -> bf16 conversion
// ---------------------------------------------------------------------------
__global__ __launch_bounds__(256) void cvt_kernel(const float* __restrict__ src,
                                                  __hip_bfloat16* __restrict__ dst, int n) {
  int i = blockIdx.x * 256 + threadIdx.x;
  if (i < n) dst[i] = __float2bfloat16(src[i]);
}

// ---------------------------------------------------------------------------
// Kernel 1: QKV projection GEMM.  y = x @ W^T + b.
// One wave computes a 16x64 output tile of one of {Q,K,V}; the k-loop is
// software-pipelined (next A + 4 B fragments prefetched during WMMAs).
// Q,K stored [bh, s, 64] bf16 ; V stored transposed [bh, 64, s] bf16.
// ---------------------------------------------------------------------------
__global__ __launch_bounds__(256) void qkv_gemm_kernel(
    const __hip_bfloat16* __restrict__ xb, const __hip_bfloat16* __restrict__ wb,
    const float* __restrict__ bq, const float* __restrict__ bk, const float* __restrict__ bv,
    __hip_bfloat16* __restrict__ Qh, __hip_bfloat16* __restrict__ Kh,
    __hip_bfloat16* __restrict__ Vt) {
  int wid = threadIdx.x >> 5, lane = threadIdx.x & 31;
  int gw = blockIdx.x * 8 + wid;                 // 9216 waves total
  int w   = gw / (256 * 12);                     // 0=Q 1=K 2=V
  int rem = gw % (256 * 12);
  int m0 = (rem / 12) * 16;
  int n0 = (rem % 12) * 64;

  f32x8 zero = {0.f, 0.f, 0.f, 0.f, 0.f, 0.f, 0.f, 0.f};
  f32x8 acc[4] = {zero, zero, zero, zero};

  const __hip_bfloat16* A  = xb + (size_t)m0 * HID_;
  const __hip_bfloat16* Wm = wb + (size_t)w * HID_ * HID_;

  // prologue: fragments for k0 = 0
  bf16x16 a_cur = load_a(A, HID_, 0);
  bf16x16 b_cur[4];
#pragma unroll
  for (int nt = 0; nt < 4; ++nt)
    b_cur[nt] = load_b(Wm + (size_t)(n0 + nt * 16) * HID_, HID_, 0);

  for (int k0 = 0; k0 < HID_; k0 += 32) {
    bf16x16 a_nxt;
    bf16x16 b_nxt[4];
    if (k0 + 32 < HID_) {                        // prefetch next k-step
      a_nxt = load_a(A, HID_, k0 + 32);
#pragma unroll
      for (int nt = 0; nt < 4; ++nt)
        b_nxt[nt] = load_b(Wm + (size_t)(n0 + nt * 16) * HID_, HID_, k0 + 32);
    }
#pragma unroll
    for (int nt = 0; nt < 4; ++nt)
      acc[nt] = wmma_bf16(a_cur, b_cur[nt], acc[nt]);
    a_cur = a_nxt;
#pragma unroll
    for (int nt = 0; nt < 4; ++nt) b_cur[nt] = b_nxt[nt];
  }

  const float* bias = (w == 0) ? bq : ((w == 1) ? bk : bv);
  int col = lane & 15, hh = lane >> 4;
#pragma unroll
  for (int nt = 0; nt < 4; ++nt) {
    int n = n0 + nt * 16 + col;
    float bval = bias[n];
    int head = n >> 6, d = n & 63;
#pragma unroll
    for (int v = 0; v < 8; ++v) {
      int m = m0 + v + 8 * hh;
      int bidx = m >> 11, s = m & (S_ - 1);
      int bh = bidx * NH_ + head;
      __hip_bfloat16 bv16 = __float2bfloat16(acc[nt][v] + bval);
      if (w == 2)      Vt[((size_t)bh * HD_ + d) * S_ + s] = bv16;
      else if (w == 0) Qh[((size_t)bh * S_ + s) * HD_ + d] = bv16;
      else             Kh[((size_t)bh * S_ + s) * HD_ + d] = bv16;
    }
  }
}

// ---------------------------------------------------------------------------
// Kernel 2: fused attention with relative key/query position bias.
// One wave per (bh, 16-row q tile); online softmax over 32-key steps.
// Bias terms computed as banded WMMA GEMMs against a dist_emb window:
//   G[l,j] = q[l] . E[j0+j],  H[r,j] = k[r] . E[j0+j],  j = (l-r)+15.
// LDS scratch is wave-private; ordering via s_wait_dscnt (no block barriers).
// ---------------------------------------------------------------------------
__global__ __launch_bounds__(256) void attn_kernel(
    const __hip_bfloat16* __restrict__ Qh, const __hip_bfloat16* __restrict__ Kh,
    const __hip_bfloat16* __restrict__ Vt, const __hip_bfloat16* __restrict__ eb,
    float* __restrict__ out) {
  __shared__ __align__(16) float          sG[8][16][32];
  __shared__ __align__(16) float          sH[8][16][32];
  __shared__ __align__(16) __hip_bfloat16 sP[8][16][32];

  int wid = threadIdx.x >> 5, lane = threadIdx.x & 31;
  int gw = blockIdx.x * 8 + wid;                 // 3072 waves
  int bh = gw >> 7;
  int l0 = (gw & 127) << 4;
  int rl = lane & 15, hq = lane >> 4;

  const __hip_bfloat16* Qb = Qh + (size_t)bh * S_ * HD_;
  const __hip_bfloat16* Kb = Kh + (size_t)bh * S_ * HD_;
  const __hip_bfloat16* Vb = Vt + (size_t)bh * HD_ * S_;

  bf16x16 aq0 = load_a(Qb + (size_t)l0 * HD_, HD_, 0);
  bf16x16 aq1 = load_a(Qb + (size_t)l0 * HD_, HD_, 32);

  f32x8 zero = {0.f, 0.f, 0.f, 0.f, 0.f, 0.f, 0.f, 0.f};
  f32x8 o[4] = {zero, zero, zero, zero};
  float mI[8], lI[8];
#pragma unroll
  for (int v = 0; v < 8; ++v) { mI[v] = -3.0e38f; lI[v] = 0.f; }

  for (int r0 = 0; r0 < S_; r0 += 32) {
    // V fragments for this 32-key step: issue early, consumed at the end.
    bf16x16 bvf[4];
#pragma unroll
    for (int dt = 0; dt < 4; ++dt)
      bvf[dt] = load_b(Vb + (size_t)(dt * 16) * S_ + r0, S_, 0);

    f32x8 Ssub[2];
#pragma unroll
    for (int sub = 0; sub < 2; ++sub) {
      int r0s = r0 + sub * 16;
      int j0 = l0 - r0s + 2032;   // E row = l - r + 2047 = j0 + j

      // group ALL global fragment loads of this subtile before the WMMA burst
      bf16x16 bk0 = load_b(Kb + (size_t)r0s * HD_, HD_, 0);
      bf16x16 bk1 = load_b(Kb + (size_t)r0s * HD_, HD_, 32);
      bf16x16 be00 = load_bE(eb, j0, 0, 0);
      bf16x16 be01 = load_bE(eb, j0, 0, 32);
      bf16x16 be10 = load_bE(eb, j0, 1, 0);
      bf16x16 be11 = load_bE(eb, j0, 1, 32);
      bf16x16 ak0 = load_a(Kb + (size_t)r0s * HD_, HD_, 0);
      bf16x16 ak1 = load_a(Kb + (size_t)r0s * HD_, HD_, 32);

      // scores = q . k^T
      f32x8 s = zero;
      s = wmma_bf16(aq0, bk0, s);
      s = wmma_bf16(aq1, bk1, s);

      // banded bias GEMMs (E-window B fragments shared between G and H)
      f32x8 g0 = zero, g1 = zero, h0 = zero, h1 = zero;
      g0 = wmma_bf16(aq0, be00, g0); g0 = wmma_bf16(aq1, be01, g0);
      g1 = wmma_bf16(aq0, be10, g1); g1 = wmma_bf16(aq1, be11, g1);
      h0 = wmma_bf16(ak0, be00, h0); h0 = wmma_bf16(ak1, be01, h0);
      h1 = wmma_bf16(ak0, be10, h1); h1 = wmma_bf16(ak1, be11, h1);

      // stage G/H tiles in wave-private LDS, gather the j = l-r+15 diagonal
#pragma unroll
      for (int v = 0; v < 8; ++v) {
        int row = v + 8 * hq;                    // C-layout row
        sG[wid][row][rl]      = g0[v];
        sG[wid][row][rl + 16] = g1[v];
        sH[wid][row][rl]      = h0[v];
        sH[wid][row][rl + 16] = h1[v];
      }
      lds_fence_wave();
#pragma unroll
      for (int v = 0; v < 8; ++v) {
        int ll = v + 8 * hq;                     // local l of this score elem
        int j = ll - rl + 15;                    // 0..30
        s[v] += sG[wid][ll][j] + sH[wid][rl][j];
      }
      Ssub[sub] = s;
    }

    // online softmax over these 32 columns (scale 1/sqrt(64) = 0.125)
#pragma unroll
    for (int v = 0; v < 8; ++v) {
      float a0 = Ssub[0][v] * 0.125f;
      float a1 = Ssub[1][v] * 0.125f;
      float mx = fmaxf(a0, a1);
      mx = fmaxf(mx, __shfl_xor(mx, 1, 32));
      mx = fmaxf(mx, __shfl_xor(mx, 2, 32));
      mx = fmaxf(mx, __shfl_xor(mx, 4, 32));
      mx = fmaxf(mx, __shfl_xor(mx, 8, 32));
      float mN = fmaxf(mI[v], mx);
      float sc = __expf(mI[v] - mN);
      float p0 = __expf(a0 - mN);
      float p1 = __expf(a1 - mN);
      float rs = p0 + p1;
      rs += __shfl_xor(rs, 1, 32);
      rs += __shfl_xor(rs, 2, 32);
      rs += __shfl_xor(rs, 4, 32);
      rs += __shfl_xor(rs, 8, 32);
      lI[v] = lI[v] * sc + rs;
      mI[v] = mN;
      o[0][v] *= sc; o[1][v] *= sc; o[2][v] *= sc; o[3][v] *= sc;
      int row = v + 8 * hq;
      sP[wid][row][rl]      = __float2bfloat16(p0);
      sP[wid][row][rl + 16] = __float2bfloat16(p1);
    }
    lds_fence_wave();

    // re-read P in A-fragment layout (row m on lanes m, m+16)
    FragU up;
    {
      const __hip_bfloat16* p = &sP[wid][rl][16 * hq];
      up.v8[0] = *reinterpret_cast<const bf16x8*>(p);
      up.v8[1] = *reinterpret_cast<const bf16x8*>(p + 8);
    }
    bf16x16 ap = up.v16;

    // O += P . V   (V transposed: column d contiguous along r)
#pragma unroll
    for (int dt = 0; dt < 4; ++dt) o[dt] = wmma_bf16(ap, bvf[dt], o[dt]);
  }

  // epilogue: normalize and store f32 output [b, s, h*64 + d]
  int bidx = bh / NH_, head = bh % NH_;
#pragma unroll
  for (int v = 0; v < 8; ++v) {
    float inv = 1.0f / lI[v];
    int srow = l0 + v + 8 * hq;
    size_t base = ((size_t)bidx * S_ + srow) * HID_ + head * HD_ + rl;
#pragma unroll
    for (int dt = 0; dt < 4; ++dt) out[base + (size_t)dt * 16] = o[dt][v] * inv;
  }
}

// ---------------------------------------------------------------------------
// Host launcher
// ---------------------------------------------------------------------------
extern "C" void kernel_launch(void* const* d_in, const int* in_sizes, int n_in,
                              void* d_out, int out_size, void* d_ws, size_t ws_size,
                              hipStream_t stream) {
  (void)in_sizes; (void)n_in; (void)out_size;
  const float* hs = (const float*)d_in[0];
  const float* Wq = (const float*)d_in[1];
  const float* bq = (const float*)d_in[2];
  const float* Wk = (const float*)d_in[3];
  const float* bk = (const float*)d_in[4];
  const float* Wv = (const float*)d_in[5];
  const float* bv = (const float*)d_in[6];
  const float* de = (const float*)d_in[7];
  float* out = (float*)d_out;

  char* ws = (char*)d_ws;
  size_t off = 0;
  const size_t nHS = (size_t)B_ * S_ * HID_;        // 3,145,728
  const size_t nW  = (size_t)HID_ * HID_;           // 589,824
  const size_t nE  = (size_t)EROWS * HD_;           // 262,080
  const size_t nQ  = (size_t)B_ * NH_ * S_ * HD_;   // 3,145,728

  auto take = [&](size_t bytes) -> char* {
    char* p = ws + off;
    off += (bytes + 255) & ~(size_t)255;
    return p;
  };
  __hip_bfloat16* xb = (__hip_bfloat16*)take(nHS * 2);
  __hip_bfloat16* wb = (__hip_bfloat16*)take(3 * nW * 2);
  __hip_bfloat16* eb = (__hip_bfloat16*)take(nE * 2);
  __hip_bfloat16* Qh = (__hip_bfloat16*)take(nQ * 2);
  __hip_bfloat16* Kh = (__hip_bfloat16*)take(nQ * 2);
  __hip_bfloat16* Vt = (__hip_bfloat16*)take(nQ * 2);
  if (off > ws_size) return;  // insufficient scratch

  cvt_kernel<<<(int)((nHS + 255) / 256), 256, 0, stream>>>(hs, xb, (int)nHS);
  cvt_kernel<<<(int)((nW + 255) / 256), 256, 0, stream>>>(Wq, wb, (int)nW);
  cvt_kernel<<<(int)((nW + 255) / 256), 256, 0, stream>>>(Wk, wb + nW, (int)nW);
  cvt_kernel<<<(int)((nW + 255) / 256), 256, 0, stream>>>(Wv, wb + 2 * nW, (int)nW);
  cvt_kernel<<<(int)((nE + 255) / 256), 256, 0, stream>>>(de, eb, (int)nE);

  // 3 matrices * 256 m-tiles * 12 n64-tiles = 9216 waves / 8 per block
  qkv_gemm_kernel<<<1152, 256, 0, stream>>>(xb, wb, bq, bk, bv, Qh, Kh, Vt);

  // 24 bh * 128 l-tiles = 3072 waves / 8 per block
  attn_kernel<<<384, 256, 0, stream>>>(Qh, Kh, Vt, eb, out);
}
